// CrossAttentionLayer_55448027791722
// MI455X (gfx1250) — matrix operand order repaired and checked by reading, not compiled, not run on hardware
//
#include <hip/hip_runtime.h>

// ---------------------------------------------------------------------------
// CDNA5 (gfx1250, wave32) cross-attention layer using v_wmma_f32_16x16x32_bf16
// ---------------------------------------------------------------------------

typedef __bf16 bf16_t;
typedef __attribute__((ext_vector_type(16))) __bf16 v16bf;
typedef __attribute__((ext_vector_type(8)))  __bf16 v8bf;
typedef __attribute__((ext_vector_type(8)))  float  v8f;
typedef __attribute__((ext_vector_type(4)))  float  v4f;
typedef __attribute__((ext_vector_type(4)))  unsigned int v4u;

#define DIM    1024
#define HEADS  16
#define HDIM   64
#define BB     2
#define NN     2048
#define BN     4096          /* B*N rows per stream */
#define K3     3072
#define SCALEF 0.125f        /* 64^-0.5 */
#define LN_EPS 1e-5f

// ---- bf16 helpers ---------------------------------------------------------
// Single v_perm_b32: pack the two high halves of (lo, hi) -> 2 x bf16.
__device__ __forceinline__ unsigned pack2(float lo, float hi) {
  return __builtin_amdgcn_perm(__float_as_uint(hi), __float_as_uint(lo),
                               0x07060302u);
}
__device__ __forceinline__ unsigned short f2bf(float f) {   // RNE, cold paths
  unsigned u = __float_as_uint(f);
  u += 0x7FFFu + ((u >> 16) & 1u);
  return (unsigned short)(u >> 16);
}
__device__ __forceinline__ v16bf cat8(v8bf lo, v8bf hi) {
  return __builtin_shufflevector(lo, hi, 0,1,2,3,4,5,6,7,8,9,10,11,12,13,14,15);
}
__device__ __forceinline__ void store8_bf(bf16_t* p, const float* v) { // 16B store
  union { v4u q; unsigned u[4]; } s;
  #pragma unroll
  for (int i = 0; i < 4; ++i) s.u[i] = pack2(v[2*i], v[2*i+1]);
  *reinterpret_cast<v4u*>(p) = s.q;
}
__device__ __forceinline__ v8f wmma_bf(v16bf a, v16bf b, v8f c) {
  return __builtin_amdgcn_wmma_f32_16x16x32_bf16(false, a, false, b, (short)0, c,
                                                 false, false);
}

// ---------------------------------------------------------------------------
// Kernel 0a: elementwise fp32 -> bf16 convert (8 elements / thread, 16B stores)
// ---------------------------------------------------------------------------
__global__ __launch_bounds__(256) void cvt_kernel(
    const float* __restrict__ src, bf16_t* __restrict__ dst, int total8) {
  int i = blockIdx.x * 256 + threadIdx.x;
  if (i >= total8) return;
  const float* p = src + (size_t)i * 8;
  v4f f0 = *reinterpret_cast<const v4f*>(p);
  v4f f1 = *reinterpret_cast<const v4f*>(p + 4);
  union { v4u q; unsigned u[4]; } s;
  s.u[0] = pack2(f0[0], f0[1]);
  s.u[1] = pack2(f0[2], f0[3]);
  s.u[2] = pack2(f1[0], f1[1]);
  s.u[3] = pack2(f1[2], f1[3]);
  *reinterpret_cast<v4u*>(dst + (size_t)i * 8) = s.q;
}

// ---------------------------------------------------------------------------
// Kernel 0b: transpose + fp32->bf16 convert of a weight matrix.
// src is [R x C] row-major fp32, dst is [C x R] row-major bf16.
// ---------------------------------------------------------------------------
__global__ __launch_bounds__(256) void transpose_cvt_kernel(
    const float* __restrict__ src, bf16_t* __restrict__ dst, int R, int C) {
  size_t i = (size_t)blockIdx.x * 256 + threadIdx.x;
  size_t total = (size_t)R * C;
  if (i >= total) return;
  size_t cc = i / (size_t)R;     // dst row  (original column)
  size_t rr = i % (size_t)R;     // dst col  (original row)
  union { unsigned short s; __bf16 b; } u;
  u.s = f2bf(src[rr * (size_t)C + cc]);
  dst[i] = u.b;
}

// ---------------------------------------------------------------------------
// Kernel 1: QKV projection. One wave = 2x2 macro-tile (two 16-wide j tiles x
// two 16-row n tiles), 4 resident accumulators, K-loop of 32: per k-step
// 8 x b128 loads feed 4 WMMAs (reuse in both directions).
// Q/K tiles (jb < 2048): transposed orientation (C[j,n])
//   -> contiguous stores into Q/K [stream][B*N][DIM] bf16.
// V tiles (jb >= 2048): standard orientation (C[n,j])
//   -> contiguous stores into Vt [stream][b][h][d][N] bf16.
// Segment boundaries (1024/2048) are multiples of 32, so a tile-pair never
// straddles Q/K/V segments; a 32-row pair never straddles a batch.
// ---------------------------------------------------------------------------
__global__ __launch_bounds__(256) void qkv_kernel(
    const bf16_t* __restrict__ Xbf, const float* __restrict__ bqkv,
    const bf16_t* __restrict__ Wt,
    bf16_t* __restrict__ Qb, bf16_t* __restrict__ Kb, bf16_t* __restrict__ Vt) {
  const int lane = threadIdx.x & 31;
  int wid = blockIdx.x * 8 + (threadIdx.x >> 5);
  const int nt = wid & 127;  wid >>= 7;            // 128 row-tile pairs (32 rows)
  const int jt = wid % 96;   wid /= 96;            // 96 col tile-pairs (3072/32)
  const int eidx = wid;                            // stream 0/1
  const int l  = lane & 15;
  const int g  = lane >> 4;
  const int jb = jt << 5;                          // first j of the pair
  const int row0 = nt << 5;                        // first row of the pair

  const bf16_t* x0 = Xbf + (size_t)(eidx * BN + row0 + l) * DIM;
  const bf16_t* x1 = x0 + (size_t)16 * DIM;
  const bf16_t* w0 = Wt + (size_t)(jb + l) * DIM;
  const bf16_t* w1 = w0 + (size_t)16 * DIM;

  v8f acc[2][2];
  #pragma unroll
  for (int i = 0; i < 2; ++i)
    #pragma unroll
    for (int j = 0; j < 2; ++j) acc[i][j] = (v8f){};

  if (jb < 2 * DIM) {
    // --- transposed orientation: acc[jq][nq] = C[j_local, n] ---------------
    for (int cb = 0; cb < DIM; cb += 32) {
      v16bf b0 = *reinterpret_cast<const v16bf*>(x0 + cb + 16 * g);
      v16bf b1 = *reinterpret_cast<const v16bf*>(x1 + cb + 16 * g);
      v16bf a0 = *reinterpret_cast<const v16bf*>(w0 + cb + 16 * g);
      v16bf a1 = *reinterpret_cast<const v16bf*>(w1 + cb + 16 * g);
      acc[0][0] = wmma_bf(a0, b0, acc[0][0]);
      acc[0][1] = wmma_bf(a0, b1, acc[0][1]);
      acc[1][0] = wmma_bf(a1, b0, acc[1][0]);
      acc[1][1] = wmma_bf(a1, b1, acc[1][1]);
    }
    bf16_t* dst; int col0;
    if (jb < DIM) { dst = Qb; col0 = jb; } else { dst = Kb; col0 = jb - DIM; }
    #pragma unroll
    for (int jq = 0; jq < 2; ++jq) {
      #pragma unroll
      for (int nq = 0; nq < 2; ++nq) {
        bf16_t* drow = dst +
            ((size_t)(eidx * BN + row0 + 16 * nq + l)) * DIM + col0 + 16 * jq + 8 * g;
        float v[8];
        #pragma unroll
        for (int r = 0; r < 8; ++r)
          v[r] = acc[jq][nq][r] + bqkv[jb + 16 * jq + 8 * g + r];
        store8_bf(drow, v);
      }
    }
  } else {
    // --- standard orientation: acc[nq][jq] = C[n_local, j] -----------------
    for (int cb = 0; cb < DIM; cb += 32) {
      v16bf a0 = *reinterpret_cast<const v16bf*>(x0 + cb + 16 * g);
      v16bf a1 = *reinterpret_cast<const v16bf*>(x1 + cb + 16 * g);
      v16bf b0 = *reinterpret_cast<const v16bf*>(w0 + cb + 16 * g);
      v16bf b1 = *reinterpret_cast<const v16bf*>(w1 + cb + 16 * g);
      acc[0][0] = wmma_bf(a0, b0, acc[0][0]);
      acc[0][1] = wmma_bf(a0, b1, acc[0][1]);
      acc[1][0] = wmma_bf(a1, b0, acc[1][0]);
      acc[1][1] = wmma_bf(a1, b1, acc[1][1]);
    }
    const int bidx = row0 >> 11;                // batch of this row pair
    const size_t vbase = (size_t)(eidx * BB + bidx) * HEADS;
    #pragma unroll
    for (int nq = 0; nq < 2; ++nq) {
      const int n0 = (row0 & (NN - 1)) + 16 * nq + 8 * g;
      #pragma unroll
      for (int jq = 0; jq < 2; ++jq) {
        const int vcol = jb - 2 * DIM + 16 * jq + l;   // h*64 + d
        const int h = vcol >> 6, d = vcol & 63;
        const float bias = bqkv[jb + 16 * jq + l];
        float v[8];
        #pragma unroll
        for (int r = 0; r < 8; ++r) v[r] = acc[nq][jq][r] + bias;
        store8_bf(Vt + ((vbase + h) * HDIM + d) * NN + n0, v);
      }
    }
  }
}

// ---------------------------------------------------------------------------
// Kernel 2: flash cross-attention. One wave = 16 queries x full head (D=64).
// Scores computed transposed (St = K Q^T), accumulators hold O^T = V^T P^T,
// so softmax row-stats live in a single lane (+ one shfl_xor(16)).
// ---------------------------------------------------------------------------
__global__ __launch_bounds__(256) void attn_kernel(
    const bf16_t* __restrict__ Qb, const bf16_t* __restrict__ Kb,
    const bf16_t* __restrict__ Vt, bf16_t* __restrict__ At) {
  const int lane = threadIdx.x & 31;
  int wid = blockIdx.x * 8 + (threadIdx.x >> 5);
  const int nt = wid & 127; wid >>= 7;          // query tile (N/16)
  const int h  = wid & 15;  wid >>= 4;
  const int b  = wid & 1;   wid >>= 1;
  const int e  = wid;                           // query stream
  const int ke = 1 - e;                         // kv stream (cross attention)
  const int l = lane & 15, g = lane >> 4;

  // Q as B-operand (two 32-wide k chunks over d)
  const bf16_t* qrow = Qb + ((size_t)(e * BN + b * NN + nt * 16 + l)) * DIM + h * HDIM;
  const v16bf qb0 = *reinterpret_cast<const v16bf*>(qrow + 16 * g);
  const v16bf qb1 = *reinterpret_cast<const v16bf*>(qrow + 32 + 16 * g);

  const bf16_t* Kbase = Kb + ((size_t)(ke * BN + b * NN)) * DIM + h * HDIM;
  const bf16_t* Vbase = Vt + ((size_t)((ke * BB + b) * HEADS + h)) * HDIM * NN;

  float m_i = -3.0e38f, l_i = 0.f;
  v8f acc[4];
  #pragma unroll
  for (int i = 0; i < 4; ++i) acc[i] = (v8f){};

  for (int sb = 0; sb < NN; sb += 32) {
    // ---- scores St[s, n] for two 16-s tiles, k = d (2 chunks of 32) -------
    const bf16_t* k0 = Kbase + (size_t)(sb + l) * DIM;
    const bf16_t* k1 = Kbase + (size_t)(sb + 16 + l) * DIM;
    v8f st0 = {}, st1 = {};
    st0 = wmma_bf(*reinterpret_cast<const v16bf*>(k0 + 16*g), qb0, st0);
    st0 = wmma_bf(*reinterpret_cast<const v16bf*>(k0 + 32 + 16*g), qb1, st0);
    st1 = wmma_bf(*reinterpret_cast<const v16bf*>(k1 + 16*g), qb0, st1);
    st1 = wmma_bf(*reinterpret_cast<const v16bf*>(k1 + 32 + 16*g), qb1, st1);

    // ---- online softmax (per lane = one softmax row n) --------------------
    float mx = st0[0];
    #pragma unroll
    for (int r = 1; r < 8; ++r) mx = fmaxf(mx, st0[r]);
    #pragma unroll
    for (int r = 0; r < 8; ++r) mx = fmaxf(mx, st1[r]);
    mx = fmaxf(mx, __shfl_xor(mx, 16, 32));
    const float m_new = fmaxf(m_i, mx * SCALEF);
    const float cf = __expf(m_i - m_new);
    float p0[8], p1[8], ls = 0.f;
    #pragma unroll
    for (int r = 0; r < 8; ++r) { p0[r] = __expf(st0[r] * SCALEF - m_new); ls += p0[r]; }
    #pragma unroll
    for (int r = 0; r < 8; ++r) { p1[r] = __expf(st1[r] * SCALEF - m_new); ls += p1[r]; }
    ls += __shfl_xor(ls, 16, 32);
    l_i = l_i * cf + ls;
    m_i = m_new;
    #pragma unroll
    for (int i = 0; i < 4; ++i) acc[i] = acc[i] * cf;

    // ---- assemble P^T B-operand (k = 32 s-values) via packed shfls --------
    unsigned o0[4], o1[4];
    #pragma unroll
    for (int i = 0; i < 4; ++i) {
      o0[i] = pack2(p0[2*i], p0[2*i+1]);
      o1[i] = pack2(p1[2*i], p1[2*i+1]);
    }
    union { v16bf v; unsigned u[8]; } pb;
    #pragma unroll
    for (int i = 0; i < 4; ++i) {
      unsigned x0 = (unsigned)__shfl_xor((int)o0[i], 16, 32);
      unsigned x1 = (unsigned)__shfl_xor((int)o1[i], 16, 32);
      if (g == 0) { pb.u[i] = o0[i]; pb.u[4 + i] = x0; }
      else        { pb.u[i] = x1;    pb.u[4 + i] = o1[i]; }
    }

    // ---- O^T[d, n] += V^T tile (A) x P^T (B), 4 d-tiles -------------------
    #pragma unroll
    for (int dt = 0; dt < 4; ++dt) {
      const bf16_t* vr = Vbase + (size_t)(16 * dt + l) * NN + sb;
      v16bf va = *reinterpret_cast<const v16bf*>(vr + 16 * g);
      acc[dt] = wmma_bf(va, pb.v, acc[dt]);
    }
  }

  // ---- finalize: O[n, d] = O^T / l, contiguous 16B stores -----------------
  const float inv = 1.f / l_i;
  bf16_t* orow = At + ((size_t)(e * BN + b * NN + nt * 16 + l)) * DIM + h * HDIM + 8 * g;
  #pragma unroll
  for (int dt = 0; dt < 4; ++dt) {
    float tv[8];
    #pragma unroll
    for (int r = 0; r < 8; ++r) tv[r] = acc[dt][r] * inv;
    store8_bf(orow + 16 * dt, tv);
  }
}

// ---------------------------------------------------------------------------
// Kernel 3: output projection + bias + residual + LayerNorm.
// Block = 16 rows x 1024 cols; wave w covers 128 cols (8 j-tiles) with the
// k-loop OUTER and 8 resident accumulators so the shared attn^T B operand is
// loaded once per k-step. Transposed orientation keeps LN row stats per-lane;
// cross-wave reduce through LDS.
// ---------------------------------------------------------------------------
__global__ __launch_bounds__(256) void proj_ln_kernel(
    const float* __restrict__ e0, const float* __restrict__ e1,
    const float* __restrict__ bout, const float* __restrict__ gamma,
    const float* __restrict__ beta, const bf16_t* __restrict__ Wt,
    const bf16_t* __restrict__ At, float* __restrict__ out) {
  __shared__ float red[8][16][2];
  const int lane = threadIdx.x & 31;
  const int w = threadIdx.x >> 5;
  const int e  = blockIdx.x >> 8;
  const int nt = blockIdx.x & 255;
  const int l = lane & 15, g = lane >> 4;
  const float* X = e ? e1 : e0;
  const size_t row = (size_t)(nt * 16 + l);                 // global row 0..4095
  const bf16_t* arow  = At + ((size_t)(e * BN) + row) * DIM; // attn^T B-operand
  const bf16_t* wbase = Wt + (size_t)(w * 128 + l) * DIM;    // A rows j

  v8f c8[8];
  #pragma unroll
  for (int jt8 = 0; jt8 < 8; ++jt8) c8[jt8] = (v8f){};

  for (int cb = 0; cb < DIM; cb += 32) {
    v16bf bmat = *reinterpret_cast<const v16bf*>(arow + cb + 16 * g);
    #pragma unroll
    for (int jt8 = 0; jt8 < 8; ++jt8) {
      const bf16_t* wr = wbase + (size_t)jt8 * 16 * DIM + cb + 16 * g;
      v16bf a = *reinterpret_cast<const v16bf*>(wr);
      c8[jt8] = wmma_bf(a, bmat, c8[jt8]);
    }
  }

  float vals[8][8];
  float sum = 0.f, sumsq = 0.f;
  #pragma unroll
  for (int jt8 = 0; jt8 < 8; ++jt8) {
    const int j0 = w * 128 + jt8 * 16 + 8 * g;
    const float* bp = bout + j0;
    const float* xp = X + row * DIM + j0;
    #pragma unroll
    for (int r = 0; r < 8; ++r) {
      float v = c8[jt8][r] + bp[r] + xp[r];
      vals[jt8][r] = v;
      sum += v;
      sumsq += v * v;
    }
  }
  sum   += __shfl_xor(sum, 16, 32);
  sumsq += __shfl_xor(sumsq, 16, 32);
  if (g == 0) { red[w][l][0] = sum; red[w][l][1] = sumsq; }
  __syncthreads();
  float ts = 0.f, tq = 0.f;
  #pragma unroll
  for (int i = 0; i < 8; ++i) { ts += red[i][l][0]; tq += red[i][l][1]; }
  const float mu  = ts * (1.f / DIM);
  const float var = tq * (1.f / DIM) - mu * mu;
  const float rs  = rsqrtf(var + LN_EPS);

  float* orow = out + (size_t)e * BN * DIM + row * DIM;
  #pragma unroll
  for (int jt8 = 0; jt8 < 8; ++jt8) {
    const int j0 = w * 128 + jt8 * 16 + 8 * g;
    #pragma unroll
    for (int r = 0; r < 8; ++r)
      orow[j0 + r] = (vals[jt8][r] - mu) * rs * gamma[j0 + r] + beta[j0 + r];
  }
}

// ---------------------------------------------------------------------------
// Host launcher. Workspace layout (bf16 elements):
//   WqkvT [3072*1024]         @ 0
//   WoutT [1024*1024]         @ 3145728
//   Xbf   [2*4096*1024]       @ 4194304
//   Q     [2*4096*1024]       @ 12582912
//   K     [2*4096*1024]       @ 20971520
//   Vt    [2*2*16*64*2048]    @ 29360128
//   attn  [2*4096*1024]       @ 37748736   (end = 46137344 bf16 = ~88 MB)
// ---------------------------------------------------------------------------
extern "C" void kernel_launch(void* const* d_in, const int* in_sizes, int n_in,
                              void* d_out, int out_size, void* d_ws, size_t ws_size,
                              hipStream_t stream) {
  (void)in_sizes; (void)n_in; (void)out_size; (void)ws_size;
  const float* emb1  = (const float*)d_in[0];
  const float* emb2  = (const float*)d_in[1];
  const float* Wqkv  = (const float*)d_in[2];
  const float* bqkv  = (const float*)d_in[3];
  const float* Wout  = (const float*)d_in[4];
  const float* bout  = (const float*)d_in[5];
  const float* gamma = (const float*)d_in[6];
  const float* beta  = (const float*)d_in[7];
  float* out = (float*)d_out;

  bf16_t* ws    = (bf16_t*)d_ws;
  bf16_t* WqkvT = ws;
  bf16_t* WoutT = ws + 3145728;
  bf16_t* Xbf   = ws + 4194304;
  bf16_t* Qb    = ws + 12582912;
  bf16_t* Kb    = ws + 20971520;
  bf16_t* Vt    = ws + 29360128;
  bf16_t* At    = ws + 37748736;

  transpose_cvt_kernel<<<12288, 256, 0, stream>>>(Wqkv, WqkvT, DIM, K3);
  transpose_cvt_kernel<<<4096,  256, 0, stream>>>(Wout, WoutT, DIM, DIM);
  // 4194304 elements per stream / 8 per thread / 256 per block = 2048 blocks
  cvt_kernel<<<2048, 256, 0, stream>>>(emb1, Xbf,           524288);
  cvt_kernel<<<2048, 256, 0, stream>>>(emb2, Xbf + 4194304, 524288);
  // 2 streams * 96 j-tile-pairs * 128 n-tile-pairs = 24576 waves / 8 per block
  qkv_kernel<<<3072, 256, 0, stream>>>(Xbf, bqkv, WqkvT, Qb, Kb, Vt);
  // 2 * 2 * 16 * 128 = 8192 waves / 8 per block
  attn_kernel<<<1024, 256, 0, stream>>>(Qb, Kb, Vt, At);
  // 2 streams * 256 row-tiles blocks
  proj_ln_kernel<<<512, 256, 0, stream>>>(emb1, emb2, bout, gamma, beta,
                                          WoutT, At, out);
}
